// DownNet_mat_2ring_simple_75505525063940
// MI455X (gfx1250) — matrix operand order repaired and checked by reading, not compiled, not run on hardware
//
#include <hip/hip_runtime.h>
#include <hip/hip_bf16.h>
#include <math.h>

#define KNB   19
#define BATCH 8

typedef __attribute__((ext_vector_type(16))) __bf16 v16bf;
typedef __attribute__((ext_vector_type(8)))  __bf16 v8bf;
typedef __attribute__((ext_vector_type(8)))  float  v8f;
typedef __attribute__((ext_vector_type(4)))  unsigned int u32x4;
typedef __attribute__((ext_vector_type(8)))  int i32x8;
typedef __attribute__((ext_vector_type(4)))  int i32x4;

#if defined(__has_builtin)
#  if __has_builtin(__builtin_amdgcn_tensor_load_to_lds)
#    define HAVE_TDM 1
#  else
#    define HAVE_TDM 0
#  endif
#else
#  define HAVE_TDM 0
#endif

#if HAVE_TDM
// Issue a TDM 2-D tile load: rows x 32 bf16 elements, row stride strideElems,
// from gsrc into LDS byte offset ldsOff (rows stored back-to-back, 64B each).
__device__ __forceinline__ void tdm_load_tile(const unsigned short* gsrc,
                                              unsigned ldsOff, int rows,
                                              int strideElems) {
  unsigned long long ga = (unsigned long long)(size_t)gsrc;
  u32x4 g0;
  g0.x = 1u;                                            // count=1 (valid user D#)
  g0.y = ldsOff;                                        // lds_addr (bytes)
  g0.z = (unsigned)ga;                                  // global_addr[31:0]
  g0.w = (unsigned)((ga >> 32) & 0x01FFFFFFu) | 0x80000000u; // addr[56:32] | type=2
  i32x8 g1;
  g1[0] = 0x00010000;                                   // data_size=1 (2 bytes)
  g1[1] = (int)((strideElems & 0xFFFF) << 16);          // tensor_dim0[15:0]
  g1[2] = (int)(((strideElems >> 16) & 0xFFFF) |        // tensor_dim0[31:16]
                ((rows & 0xFFFF) << 16));               // tensor_dim1[15:0]
  g1[3] = (int)(((rows >> 16) & 0xFFFF) | (32 << 16));  // tensor_dim1[31:16] | tile_dim0=32
  g1[4] = (int)(rows & 0xFFFF);                         // tile_dim1=rows, tile_dim2=0
  g1[5] = strideElems;                                  // tensor_dim0_stride[31:0]
  g1[6] = 0;                                            // stride hi, dim1_stride lo
  g1[7] = 0;
  i32x4 z4 = {0, 0, 0, 0};
#if __clang_major__ >= 23
  i32x8 z8 = {0, 0, 0, 0, 0, 0, 0, 0};
  __builtin_amdgcn_tensor_load_to_lds(g0, g1, z4, z4, z8, 0);
#else
  __builtin_amdgcn_tensor_load_to_lds(g0, g1, z4, z4, 0);
#endif
}
#endif

// ---------------------------------------------------------------- utilities
__global__ void k_zero(float* __restrict__ p, int n) {
  int i = blockIdx.x * blockDim.x + threadIdx.x;
  if (i < n) p[i] = 0.f;
}

// W[KC, Cout] f32 (row-major) -> Wt[Cout, KCpad] bf16 bits, zero padded.
__global__ void k_prep_w(const float* __restrict__ W, unsigned short* __restrict__ Wt,
                         int KC, int KCpad, int Cout) {
  int i = blockIdx.x * blockDim.x + threadIdx.x;
  if (i >= Cout * KCpad) return;
  int n = i / KCpad;
  int k = i - n * KCpad;
  float v = (k < KC) ? W[k * Cout + n] : 0.f;
  Wt[n * KCpad + k] = (unsigned short)(__float_as_uint(v) >> 16);
}

// ------------------------------------------------- gather + GEMM via WMMA
// y[B*Vn, Cout] = gather(x, no) @ W + bias; optional per-channel sum/sumsq
// into stats[2*Cout] for BatchNorm batch statistics.
//
// Block = 128 threads (4 waves). Block computes 16 rows x 64*NT columns.
// A tile (16x32 bf16) gathered to LDS once per k-step, feeds 4*NT WMMAs.
// Weight panel (64*NT x 32 bf16) is DMAd to LDS by the Tensor Data Mover,
// double-buffered so the DMA for step kk+32 overlaps the WMMAs of step kk.
template<int NT>
__global__ void __launch_bounds__(128)
k_conv_wmma(const float* __restrict__ x, const int* __restrict__ no,
            const unsigned short* __restrict__ Wt, const float* __restrict__ bias,
            float* __restrict__ y, float* __restrict__ stats,
            int Vn, int C, int log2C, int Kn, int KC, int KCpad, int Cout)
{
  __shared__ __align__(32) unsigned short As[16 * 32];
#if HAVE_TDM
  __shared__ __align__(128) unsigned short Ws[2][NT * 64 * 32];
#endif

  const int tid  = threadIdx.x;
  const int lane = tid & 31;
  const int half = (tid >> 4) & 1;                 // K-half within wave
  const int waveId = __builtin_amdgcn_readfirstlane(tid >> 5);  // uniform -> scalar branch
  const int rowTile = blockIdx.x;
  const int colBase = blockIdx.y * (64 * NT);
  const int n = (tid >> 5) * 16 + (lane & 15);     // channel within this panel

  // ---- staging coordinates: thread tid fills row Ms, cols [t0, t0+4)
  const int Ms = tid >> 3;             // 0..15
  const int t0 = (tid & 7) << 2;       // 0,4,...,28
  const int gRowS = rowTile * 16 + Ms;
  const int b  = gRowS / Vn;
  const int vv = gRowS - b * Vn;
  const int xbase = b * Vn * C;
  const int* noRow = no ? (no + vv * Kn) : nullptr;

  const int Mrow = lane & 15;
#if !HAVE_TDM
  const unsigned short* wrow = Wt + (size_t)(colBase + n) * KCpad + 16 * half;
#endif

  v8f acc[NT];
#pragma unroll
  for (int s = 0; s < NT; ++s) acc[s] = v8f{};

#if HAVE_TDM
  const unsigned wsBase = (unsigned)(size_t)(void*)&Ws[0][0];
  const unsigned panelBytes = (unsigned)(NT * 64 * 32 * 2);
  const unsigned short* wpanel = Wt + (size_t)colBase * KCpad;
  int p = 0;
  if (waveId == 0) tdm_load_tile(wpanel, wsBase, 64 * NT, KCpad);
#endif

#pragma unroll 2
  for (int kk = 0; kk < KCpad; kk += 32) {
#if HAVE_TDM
    if (waveId == 0 && kk + 32 < KCpad)
      tdm_load_tile(wpanel + kk + 32, wsBase + (p ^ 1) * panelBytes, 64 * NT, KCpad);
#endif
    // ---- gather 16x32 A tile -> LDS (bf16, truncating convert)
    if (C >= 32) {
      // one neighbor per 32-k chunk; channels contiguous -> b128 load
      int j   = kk >> log2C;
      int ch0 = (kk & (C - 1)) + t0;
      int src = noRow ? noRow[j] : vv;
      const float4 v4 = *reinterpret_cast<const float4*>(x + xbase + src * C + ch0);
      uint2 pk;
      pk.x = (__float_as_uint(v4.x) >> 16) | (__float_as_uint(v4.y) & 0xFFFF0000u);
      pk.y = (__float_as_uint(v4.z) >> 16) | (__float_as_uint(v4.w) & 0xFFFF0000u);
      *reinterpret_cast<uint2*>(&As[(Ms << 5) + t0]) = pk;
      if (kk + 32 < KCpad) {   // hide next step's indexed-gather latency
        int jn = (kk + 32) >> log2C;
        int srcn = noRow ? noRow[jn] : vv;
        __builtin_prefetch(x + xbase + srcn * C + ((kk + 32) & (C - 1)) + t0, 0, 1);
      }
    } else {
      // generic path (level-0 conv1: C=4, KC=76 padded to 96)
#pragma unroll
      for (int e = 0; e < 4; ++e) {
        int t = t0 + e;
        int kidx = kk + t;
        float v = 0.f;
        if (kidx < KC) {
          int j  = kidx >> log2C;
          int ch = kidx & (C - 1);
          int src = noRow ? noRow[j] : vv;
          v = x[xbase + src * C + ch];
        }
        As[(Ms << 5) + t] = (unsigned short)(__float_as_uint(v) >> 16);
      }
    }
#if HAVE_TDM
    if (waveId == 0) {
      if (kk + 32 < KCpad) __builtin_amdgcn_s_wait_tensorcnt(1);  // older DMA done
      else                 __builtin_amdgcn_s_wait_tensorcnt(0);
    }
#endif
    __syncthreads();   // A tile + weight panel visible to all waves

    // A fragment (contiguous k order, matching B): lane(M, half) reads
    // K in [16*half, 16*half+16)
    union { v16bf v16; v8bf v8[2]; } a;
    a.v8[0] = *reinterpret_cast<const v8bf*>(&As[(Mrow << 5) + 16 * half]);
    a.v8[1] = *reinterpret_cast<const v8bf*>(&As[(Mrow << 5) + 16 * half + 8]);

    // NT column sub-tiles reuse the same A fragment
#pragma unroll
    for (int s = 0; s < NT; ++s) {
#if HAVE_TDM
      v16bf bw = *reinterpret_cast<const v16bf*>(&Ws[p][(n + s * 64) * 32 + 16 * half]);
#else
      v16bf bw = *reinterpret_cast<const v16bf*>(wrow + s * 64 * KCpad + kk);
#endif
      acc[s] = __builtin_amdgcn_wmma_f32_16x16x32_bf16(
          /*neg_a=*/false, a.v16, /*neg_b=*/false, bw,
          /*c_mod=*/(short)0, acc[s], /*reuse_a=*/false, /*reuse_b=*/false);
    }
    __syncthreads();   // all waves done reading before next stage overwrites
#if HAVE_TDM
    p ^= 1;
#endif
  }

  // C/D layout: lane holds column n, rows M = r + 8*half
#pragma unroll
  for (int s = 0; s < NT; ++s) {
    const int ns = colBase + n + s * 64;
    const float bb = bias[ns];
    float sum = 0.f, sum2 = 0.f;
#pragma unroll
    for (int r = 0; r < 8; ++r) {
      int gRow = rowTile * 16 + r + 8 * half;
      float v = acc[s][r] + bb;
      y[gRow * Cout + ns] = v;
      sum += v; sum2 += v * v;
    }
    if (stats) {
      atomicAdd(&stats[ns], sum);
      atomicAdd(&stats[Cout + ns], sum2);
    }
  }
}

// ------------------------------------------- BatchNorm (batch stats) + LeakyReLU
__global__ void k_bn_lrelu(float* __restrict__ y, const float* __restrict__ stats,
                           const float* __restrict__ g, const float* __restrict__ be,
                           int Nrows, int Cout) {
  int i = blockIdx.x * blockDim.x + threadIdx.x;
  if (i >= Nrows * Cout) return;
  int n = i % Cout;
  float inv = 1.f / (float)Nrows;
  float m   = stats[n] * inv;
  float var = stats[Cout + n] * inv - m * m;   // biased variance
  float v = (y[i] - m) * rsqrtf(var + 1e-5f) * g[n] + be[n];
  y[i] = (v >= 0.f) ? v : 0.2f * v;
}

// ------------------------------------------------------- 2-ring mean pooling
__global__ void k_pool(const float* __restrict__ x, const int* __restrict__ no,
                       float* __restrict__ y, int VnIn, int Vout, int C, int Kn) {
  int i = blockIdx.x * blockDim.x + threadIdx.x;
  int tot = BATCH * Vout * C;
  if (i >= tot) return;
  int c = i % C;
  int r = i / C;
  int v = r % Vout;
  int b = r / Vout;
  const int* nr = no + v * Kn;
  int base = b * VnIn * C + c;
  float s = 0.f;
  for (int j = 0; j < Kn; ++j) s += x[base + nr[j] * C];
  y[i] = s * (1.f / (float)Kn);
}

__global__ void k_tanh(float* __restrict__ y, int n) {
  int i = blockIdx.x * blockDim.x + threadIdx.x;
  if (i < n) y[i] = tanhf(y[i]);
}

// o1[b,o] = h[b,:D] . Wo1[:,o] + bo1[o]   (D=21504, O=100)
__global__ void __launch_bounds__(128)
k_head1(const float* __restrict__ h, const float* __restrict__ W,
        const float* __restrict__ b, float* __restrict__ o, int D, int O) {
  int out = blockIdx.x;
  int bi = out / O, oi = out - bi * O;
  __shared__ float red[128];
  float s = 0.f;
  for (int k = threadIdx.x; k < D; k += 128) s += h[bi * D + k] * W[k * O + oi];
  red[threadIdx.x] = s;
  __syncthreads();
  for (int st = 64; st > 0; st >>= 1) {
    if (threadIdx.x < st) red[threadIdx.x] += red[threadIdx.x + st];
    __syncthreads();
  }
  if (threadIdx.x == 0) o[out] = red[0] + b[oi];
}

__global__ void k_head2(const float* __restrict__ o1, const float* __restrict__ W,
                        const float* __restrict__ b, float* __restrict__ out, int O) {
  int i = blockIdx.x * blockDim.x + threadIdx.x;
  if (i >= BATCH) return;
  float s = 0.f;
  for (int k = 0; k < O; ++k) s += o1[i * O + k] * W[k];
  out[i] = s + b[0];
}

// =========================================================== host launcher
extern "C" void kernel_launch(void* const* d_in, const int* in_sizes, int n_in,
                              void* d_out, int out_size, void* d_ws, size_t ws_size,
                              hipStream_t stream) {
  (void)in_sizes; (void)n_in; (void)out_size; (void)ws_size;

  static const int V[6]  = {40962, 10242, 2562, 642, 162, 42};
  static const int CH[7] = {4, 64, 64, 128, 128, 256, 512};

  const float* x_in = (const float*)d_in[0];
  const int* no[6];
  for (int i = 0; i < 6; ++i) no[i] = (const int*)d_in[1 + i];
  auto P = [&](int idx) { return (const float*)d_in[7 + idx]; };
  const float* Wf  = (const float*)d_in[55];
  const float* bf  = (const float*)d_in[56];
  const float* Wo1 = (const float*)d_in[57];
  const float* bo1 = (const float*)d_in[58];
  const float* Wo2 = (const float*)d_in[59];
  const float* bo2 = (const float*)d_in[60];

  // workspace layout
  char* ws = (char*)d_ws;
  const size_t SZ = (size_t)BATCH * 40962 * 64 * sizeof(float);   // 83.9 MB max activation
  float* bufA = (float*)ws;
  float* bufB = (float*)(ws + SZ);
  unsigned short* wt = (unsigned short*)(ws + 2 * SZ);            // bf16 weights (<= 9728*512)
  float* stats = (float*)(ws + 2 * SZ + (size_t)9728 * 512 * sizeof(unsigned short));
  float* o1 = stats + 1024;

  auto gemm = [&](const float* xin, const int* noP, int Vn, int Cin, int Kn,
                  const float* W, const float* bias, float* yout, int Cout, float* st) {
    int KC = Kn * Cin;
    int KCpad = (KC + 31) & ~31;
    int Nrows = BATCH * Vn;
    int log2C = __builtin_ctz((unsigned)Cin);
    int tot = Cout * KCpad;
    k_prep_w<<<dim3((tot + 255) / 256), dim3(256), 0, stream>>>(W, wt, KC, KCpad, Cout);
    if (st)
      k_zero<<<dim3((2 * Cout + 255) / 256), dim3(256), 0, stream>>>(st, 2 * Cout);
    int NT = Cout / 64; if (NT > 4) NT = 4;         // LDS cap: Ws = NT*8KB double-buffered
    dim3 grid(Nrows / 16, Cout / (64 * NT));
    switch (NT) {
      case 1: k_conv_wmma<1><<<grid, dim3(128), 0, stream>>>(
                  xin, noP, wt, bias, yout, st, Vn, Cin, log2C, Kn, KC, KCpad, Cout); break;
      case 2: k_conv_wmma<2><<<grid, dim3(128), 0, stream>>>(
                  xin, noP, wt, bias, yout, st, Vn, Cin, log2C, Kn, KC, KCpad, Cout); break;
      default: k_conv_wmma<4><<<grid, dim3(128), 0, stream>>>(
                  xin, noP, wt, bias, yout, st, Vn, Cin, log2C, Kn, KC, KCpad, Cout); break;
    }
  };

  const float* cur = x_in;
  float* nxt = bufA;
  float* other = bufB;
  auto swapbuf = [&]() { float* t = nxt; nxt = other; other = t; };

  for (int i = 0; i < 6; ++i) {
    int Cin = CH[i], Cout = CH[i + 1], Vn = V[i];
    if (i > 0) {
      int tot = BATCH * Vn * Cin;
      k_pool<<<dim3((tot + 255) / 256), dim3(256), 0, stream>>>(
          cur, no[i - 1], nxt, V[i - 1], Vn, Cin, KNB);
      cur = nxt; swapbuf();
    }
    // conv1 + BN + LeakyReLU
    gemm(cur, no[i], Vn, Cin, KNB, P(8 * i + 0), P(8 * i + 1), nxt, Cout, stats);
    k_bn_lrelu<<<dim3((BATCH * Vn * Cout + 255) / 256), dim3(256), 0, stream>>>(
        nxt, stats, P(8 * i + 2), P(8 * i + 3), BATCH * Vn, Cout);
    cur = nxt; swapbuf();
    // conv2 + BN + LeakyReLU
    gemm(cur, no[i], Vn, Cout, KNB, P(8 * i + 4), P(8 * i + 5), nxt, Cout, stats);
    k_bn_lrelu<<<dim3((BATCH * Vn * Cout + 255) / 256), dim3(256), 0, stream>>>(
        nxt, stats, P(8 * i + 6), P(8 * i + 7), BATCH * Vn, Cout);
    cur = nxt; swapbuf();
  }

  // per-vertex FC (512->512) + tanh : plain GEMM (no gather, no BN stats)
  gemm(cur, (const int*)nullptr, 42, 512, 1, Wf, bf, nxt, 512, (float*)nullptr);
  int nh = BATCH * 42 * 512;
  k_tanh<<<dim3((nh + 255) / 256), dim3(256), 0, stream>>>(nxt, nh);

  // head: [8, 21504] @ [21504, 100] then [8, 100] @ [100, 1]
  k_head1<<<dim3(BATCH * 100), dim3(128), 0, stream>>>(nxt, Wo1, bo1, o1, 42 * 512, 100);
  k_head2<<<dim3(1), dim3(32), 0, stream>>>(o1, Wo2, bo2, (float*)d_out, 100);
}
